// Gemma4VisionAttention_60979945669121
// MI455X (gfx1250) — matrix operand order repaired and checked by reading, not compiled, not run on hardware
//
#include <hip/hip_runtime.h>
#include <hip/hip_bf16.h>
#include <stdint.h>

#define B_   2
#define S_   2048
#define D_   1024
#define HD_  64
#define NQ_  16
#define NKV_ 4

typedef __attribute__((ext_vector_type(16))) __bf16         v16bf;
typedef __attribute__((ext_vector_type(8)))  float          v8f;
typedef __attribute__((ext_vector_type(8)))  unsigned short ushort8;

union FragAB {
  v16bf          v;
  ushort8        h8[2];
  uint32_t       d[8];
  unsigned short u[16];
};

__device__ inline unsigned short f2bf(float f) {
  union { float f; uint32_t u; } c; c.f = f;
  uint32_t u = c.u;
  return (unsigned short)((u + 0x7FFFu + ((u >> 16) & 1u)) >> 16);
}

__device__ inline v8f v8f_zero() {
  v8f z;
#pragma unroll
  for (int i = 0; i < 8; ++i) z[i] = 0.0f;
  return z;
}

// branchless 50*tanh(x/50) = 50 - 100/(exp(x*0.04)+1); logits bounded (|x|<=64)
__device__ inline float softcap50(float x) {
  float e = __expf(x * 0.04f);
  return 50.0f - 100.0f * __builtin_amdgcn_rcpf(e + 1.0f);
}

// ---------------------------------------------------------------------------
// Generic bf16 WMMA GEMM: C(f32) = A * B, A is f32 (converted) or bf16,
// B is f32 weights converted on the fly. Block = 128 thr (4 waves), each
// block computes a 64x64 tile; per-z weight slab (per attention head).
// ---------------------------------------------------------------------------
template <bool A_F32>
__global__ __launch_bounds__(128)
void gemm_bf16_wmma(const void* __restrict__ Avoid, const float* __restrict__ Bmat,
                    float* __restrict__ C, int K, int lda, int ldb, int ldc,
                    int c_col_base, int c_col_stride, long long b_z_stride)
{
  __shared__ __align__(16) unsigned short lds_a [64 * 32]; // [row][k]
  __shared__ __align__(16) unsigned short lds_bt[64 * 32]; // [n][k] (transposed)

  const int tid   = threadIdx.x;
  const int wv    = tid >> 5;
  const int lane  = tid & 31;
  const int halfw = lane >> 4;
  const int lm    = lane & 15;

  const int row0 = blockIdx.y * 64;
  const int n0   = blockIdx.x * 64;
  const int z    = blockIdx.z;
  const float* Bp = Bmat + (long long)z * b_z_stride;
  const int ccol0 = c_col_base + z * c_col_stride + n0;

  const float*          Af = (const float*)Avoid;
  const unsigned short* Ab = (const unsigned short*)Avoid;

  v8f acc[4];
#pragma unroll
  for (int t = 0; t < 4; ++t) acc[t] = v8f_zero();

  for (int k0 = 0; k0 < K; k0 += 32) {
    const bool has_next = (k0 + 32) < K;
    // stage A tile 64x32 -> bf16 LDS (+ prefetch next K block)
    for (int e = tid; e < 64 * 32; e += 128) {
      int r = e >> 5, c = e & 31;
      unsigned short bv;
      if (A_F32) {
        const float* p = &Af[(long long)(row0 + r) * lda + k0 + c];
        if (has_next) __builtin_prefetch(p + 32, 0, 3);
        bv = f2bf(*p);
      } else {
        const unsigned short* p = &Ab[(long long)(row0 + r) * lda + k0 + c];
        if (has_next) __builtin_prefetch(p + 32, 0, 3);
        bv = *p;
      }
      lds_a[r * 32 + c] = bv;
    }
    // stage B tile 32x64 transposed -> LDS [n][k] so frag reads are b128
    for (int e = tid; e < 64 * 32; e += 128) {
      int k = e >> 6, n = e & 63;
      const float* p = &Bp[(long long)(k0 + k) * ldb + n0 + n];
      if (has_next) __builtin_prefetch(p + 32LL * ldb, 0, 3);
      lds_bt[n * 32 + k] = f2bf(*p);
    }
    __syncthreads();

    // A fragment: 16-bit A 16x32 layout (K pairs contiguous -> b32 reads)
    FragAB a;
#pragma unroll
    for (int v = 0; v < 8; ++v) {
      int kb = halfw * 16 + 2 * v;
      a.d[v] = *(const uint32_t*)&lds_a[(wv * 16 + lm) * 32 + kb];
    }
#pragma unroll
    for (int t = 0; t < 4; ++t) {
      // B fragment: lane = N, element e -> K = half*16 + e (contiguous 16)
      FragAB b;
      const ushort8* p = (const ushort8*)&lds_bt[(t * 16 + lm) * 32 + halfw * 16];
      b.h8[0] = p[0];
      b.h8[1] = p[1];
      acc[t] = __builtin_amdgcn_wmma_f32_16x16x32_bf16(
          false, a.v, false, b.v, (short)0, acc[t], false, false);
    }
    __syncthreads();
  }

#pragma unroll
  for (int t = 0; t < 4; ++t)
#pragma unroll
    for (int j = 0; j < 8; ++j) {
      int row = row0 + wv * 16 + j + 8 * halfw;
      int col = ccol0 + t * 16 + lm;
      C[(long long)row * ldc + col] = acc[t][j];
    }
}

// ---------------------------------------------------------------------------
// RMS-norm (+scale) and 2-part RoPE. One wave per (b, s, head) vector of 64.
// heads 0..15 = Q (norm+rope, row-major), 16..19 = K (norm+rope, row-major),
// 20..23 = V (norm only, stored TRANSPOSED (64 x S) for the PV B-matrix).
// ---------------------------------------------------------------------------
__global__ __launch_bounds__(128)
void norm_rope_kernel(const float* __restrict__ qkv,
                      const int*   __restrict__ pos_ids,
                      const float* __restrict__ q_scale,
                      const float* __restrict__ k_scale,
                      unsigned short* __restrict__ q_bf,
                      unsigned short* __restrict__ k_bf,
                      unsigned short* __restrict__ vT_bf)
{
  const int wv   = threadIdx.x >> 5;
  const int lane = threadIdx.x & 31;
  const int vec  = blockIdx.x * 4 + wv;

  const int b   = vec / (S_ * 24);
  const int rem = vec % (S_ * 24);
  const int s   = rem / 24;
  const int hd  = rem % 24;

  const long long row = (long long)b * S_ + s;
  const float* src = qkv + row * 1536 + hd * 64;
  float x0 = src[lane];
  float x1 = src[lane + 32];

  float ss = x0 * x0 + x1 * x1;
#pragma unroll
  for (int off = 16; off >= 1; off >>= 1) ss += __shfl_xor(ss, off, 32);
  const float inv = rsqrtf(ss * (1.0f / 64.0f) + 1e-6f);

  if (hd < 20) {
    const float* sc = (hd < 16) ? q_scale : k_scale;
    float n0 = x0 * inv * (1.0f + sc[lane]);
    float n1 = x1 * inv * (1.0f + sc[lane + 32]);
    // RoPE: dims 0..31 use pos0, dims 32..63 use pos1; rotate within 32.
    float p0 = (float)pos_ids[row * 2 + 0];
    float p1 = (float)pos_ids[row * 2 + 1];
    float fr = __powf(10000.0f, -(float)(lane & 15) / 16.0f);
    float th0 = p0 * fr, th1 = p1 * fr;
    float part0 = __shfl_xor(n0, 16, 32);
    float part1 = __shfl_xor(n1, 16, 32);
    float r0 = (lane < 16) ? -part0 : part0;
    float r1 = (lane < 16) ? -part1 : part1;
    float o0 = n0 * __cosf(th0) + r0 * __sinf(th0);
    float o1 = n1 * __cosf(th1) + r1 * __sinf(th1);
    if (hd < 16) {
      unsigned short* dst = q_bf + ((long long)(b * NQ_ + hd) * S_ + s) * 64;
      dst[lane]      = f2bf(o0);
      dst[lane + 32] = f2bf(o1);
    } else {
      unsigned short* dst = k_bf + ((long long)(b * NKV_ + (hd - 16)) * S_ + s) * 64;
      dst[lane]      = f2bf(o0);
      dst[lane + 32] = f2bf(o1);
    }
  } else {
    unsigned short* dst = vT_bf + (long long)(b * NKV_ + (hd - 20)) * 64 * S_;
    dst[(long long)lane * S_ + s]        = f2bf(x0 * inv);
    dst[(long long)(lane + 32) * S_ + s] = f2bf(x1 * inv);
  }
}

// ---------------------------------------------------------------------------
// Flash-style soft-capped attention. One wave per (b, q-head, 16-query tile).
// 32 keys per iteration: 4 WMMA for logits (2 key subtiles x K=64), online
// softmax with per-row stats (D layout: lane holds rows j+8*half), P staged
// through LDS to convert D-layout -> A-layout, 4 WMMA for PV.
// ---------------------------------------------------------------------------
__global__ __launch_bounds__(32)
void attn_kernel(const unsigned short* __restrict__ q_bf,
                 const unsigned short* __restrict__ k_bf,
                 const unsigned short* __restrict__ vT_bf,
                 const unsigned char*  __restrict__ maskp,
                 unsigned short* __restrict__ attn_bf)
{
  __shared__ __align__(16) unsigned short lds_p[16 * 32]; // P tile [row][key]

  const int lane  = threadIdx.x & 31;
  const int halfw = lane >> 4;
  const int lm    = lane & 15;

  const int bid  = blockIdx.x;
  const int b    = bid / (NQ_ * (S_ / 16));
  const int rem  = bid % (NQ_ * (S_ / 16));
  const int qh   = rem / (S_ / 16);
  const int t0   = (rem % (S_ / 16)) * 16;
  const int kv   = qh / (NQ_ / NKV_);

  const unsigned short* qb = q_bf  + ((long long)(b * NQ_ + qh) * S_ + t0) * 64;
  const unsigned short* kp = k_bf  + (long long)(b * NKV_ + kv) * S_ * 64;
  const unsigned short* vT = vT_bf + (long long)(b * NKV_ + kv) * 64 * S_;
  const unsigned char*  mk = maskp + (long long)b * S_;

  // Q fragments for K-dim 0..31 and 32..63 (reused all iterations)
  FragAB qa[2];
#pragma unroll
  for (int kc = 0; kc < 2; ++kc)
#pragma unroll
    for (int v = 0; v < 8; ++v) {
      int K = kc * 32 + halfw * 16 + 2 * v;
      qa[kc].d[v] = *(const uint32_t*)&qb[lm * 64 + K];
    }

  v8f O[4];
#pragma unroll
  for (int t = 0; t < 4; ++t) O[t] = v8f_zero();
  float m_run[8], l_run[8];
#pragma unroll
  for (int j = 0; j < 8; ++j) { m_run[j] = -1.0e9f; l_run[j] = 0.0f; }

  for (int s0 = 0; s0 < S_; s0 += 32) {
    // prefetch next chunk's K and V rows (global_prefetch_b8)
    const int sp = (s0 + 32 < S_) ? (s0 + 32) : s0;
    __builtin_prefetch(&kp[(long long)(sp + lm) * 64], 0, 3);
    __builtin_prefetch(&kp[(long long)(sp + 16 + lm) * 64], 0, 3);
    __builtin_prefetch(&vT[(long long)(lane << 1) * S_ + sp], 0, 3);

    v8f acc0 = v8f_zero();   // keys s0..s0+15
    v8f acc1 = v8f_zero();   // keys s0+16..s0+31
#pragma unroll
    for (int kc = 0; kc < 2; ++kc) {
      FragAB kf;
      const ushort8* p0 = (const ushort8*)&kp[(long long)(s0 + lm) * 64 + kc * 32 + halfw * 16];
      kf.h8[0] = p0[0]; kf.h8[1] = p0[1];
      acc0 = __builtin_amdgcn_wmma_f32_16x16x32_bf16(
          false, qa[kc].v, false, kf.v, (short)0, acc0, false, false);
      const ushort8* p1 = (const ushort8*)&kp[(long long)(s0 + 16 + lm) * 64 + kc * 32 + halfw * 16];
      kf.h8[0] = p1[0]; kf.h8[1] = p1[1];
      acc1 = __builtin_amdgcn_wmma_f32_16x16x32_bf16(
          false, qa[kc].v, false, kf.v, (short)0, acc1, false, false);
    }

    const bool msk0 = mk[s0 + lm]      != 0;
    const bool msk1 = mk[s0 + 16 + lm] != 0;

    __syncthreads();
#pragma unroll
    for (int j = 0; j < 8; ++j) {
      float l0 = softcap50(acc0[j]);
      float l1 = softcap50(acc1[j]);
      l0 = msk0 ? l0 : -1.0e9f;
      l1 = msk1 ? l1 : -1.0e9f;
      float cm = fmaxf(l0, l1);
#pragma unroll
      for (int off = 8; off >= 1; off >>= 1) cm = fmaxf(cm, __shfl_xor(cm, off, 16));
      float m_new = fmaxf(m_run[j], cm);
      float corr  = __expf(m_run[j] - m_new);
      float p0    = __expf(l0 - m_new);
      float p1    = __expf(l1 - m_new);
      float rs    = p0 + p1;
#pragma unroll
      for (int off = 8; off >= 1; off >>= 1) rs += __shfl_xor(rs, off, 16);
      l_run[j] = l_run[j] * corr + rs;
      m_run[j] = m_new;
      const int mr = j + 8 * halfw;
      lds_p[mr * 32 + lm]      = f2bf(p0);
      lds_p[mr * 32 + 16 + lm] = f2bf(p1);
#pragma unroll
      for (int t = 0; t < 4; ++t) O[t][j] *= corr;
    }
    __syncthreads();

    // P as A-matrix (16x32 over the 32 keys of this chunk)
    FragAB pa;
#pragma unroll
    for (int v = 0; v < 8; ++v) {
      int K = halfw * 16 + 2 * v;
      pa.d[v] = *(const uint32_t*)&lds_p[lm * 32 + K];
    }
#pragma unroll
    for (int t = 0; t < 4; ++t) {
      FragAB vf;  // B[k=key][n=h] from V transposed -> contiguous b128
      const ushort8* pv = (const ushort8*)&vT[(long long)(t * 16 + lm) * S_ + s0 + halfw * 16];
      vf.h8[0] = pv[0]; vf.h8[1] = pv[1];
      O[t] = __builtin_amdgcn_wmma_f32_16x16x32_bf16(
          false, pa.v, false, vf.v, (short)0, O[t], false, false);
    }
  }

#pragma unroll
  for (int t = 0; t < 4; ++t)
#pragma unroll
    for (int j = 0; j < 8; ++j) {
      int row = t0 + j + 8 * halfw;
      int col = qh * 64 + t * 16 + lm;
      float o = O[t][j] * __builtin_amdgcn_rcpf(l_run[j]);
      attn_bf[(long long)(b * S_ + row) * (NQ_ * 64) + col] = f2bf(o);
    }
}

// ---------------------------------------------------------------------------
extern "C" void kernel_launch(void* const* d_in, const int* in_sizes, int n_in,
                              void* d_out, int out_size, void* d_ws, size_t ws_size,
                              hipStream_t stream) {
  (void)in_sizes; (void)n_in; (void)out_size; (void)ws_size;
  const float*         x    = (const float*)d_in[0];
  const unsigned char* mask = (const unsigned char*)d_in[1];
  const int*           pos  = (const int*)d_in[2];
  const float*         Wq   = (const float*)d_in[3];
  const float*         Wk   = (const float*)d_in[4];
  const float*         Wv   = (const float*)d_in[5];
  const float*         Wo   = (const float*)d_in[6];
  const float*         qs   = (const float*)d_in[7];
  const float*         ks   = (const float*)d_in[8];
  float*               out  = (float*)d_out;

  char* ws = (char*)d_ws;
  float*          qkv     = (float*)(ws);                        // 4096 x 1536 f32 = 25165824 B
  unsigned short* q_bf    = (unsigned short*)(ws + 25165824);    // (B,NQ,S,64) bf16 = 8388608 B
  unsigned short* k_bf    = (unsigned short*)(ws + 33554432);    // (B,NKV,S,64) bf16 = 2097152 B
  unsigned short* vT_bf   = (unsigned short*)(ws + 35651584);    // (B,NKV,64,S) bf16 = 2097152 B
  unsigned short* attn_bf = (unsigned short*)(ws + 37748736);    // 4096 x 1024 bf16 = 8388608 B

  dim3 blk(128);
  // QKV projections: per-head weight slab via grid.z; cols = head*64 in qkv
  gemm_bf16_wmma<true><<<dim3(1, 64, 16), blk, 0, stream>>>(
      x, Wq, qkv, 1024, 1024, 64, 1536, 0,    64, 65536LL);
  gemm_bf16_wmma<true><<<dim3(1, 64, 4),  blk, 0, stream>>>(
      x, Wk, qkv, 1024, 1024, 64, 1536, 1024, 64, 65536LL);
  gemm_bf16_wmma<true><<<dim3(1, 64, 4),  blk, 0, stream>>>(
      x, Wv, qkv, 1024, 1024, 64, 1536, 1280, 64, 65536LL);

  // RMS norm + RoPE (+ V transpose): B*S*24 vectors, 4 waves/block
  norm_rope_kernel<<<dim3((B_ * S_ * 24) / 4), blk, 0, stream>>>(
      qkv, pos, qs, ks, q_bf, k_bf, vT_bf);

  // Attention: one wave per (b, q-head, 16-query tile)
  attn_kernel<<<dim3(B_ * NQ_ * (S_ / 16)), dim3(32), 0, stream>>>(
      q_bf, k_bf, vT_bf, mask, attn_bf);

  // Output projection: attn (4096x1024 bf16) x Wo (1024x1024 f32) -> out f32
  gemm_bf16_wmma<false><<<dim3(16, 64, 1), blk, 0, stream>>>(
      attn_bf, Wo, out, 1024, 1024, 1024, 1024, 0, 0, 0LL);
}
